// KeyValAttention_65867618451509
// MI455X (gfx1250) — matrix operand order repaired and checked by reading, not compiled, not run on hardware
//
#include <hip/hip_runtime.h>

// MI455X / gfx1250: wave32, WMMA 16x16x32 bf16 (f32 accum).
// Attention with materialized attn output: memory-bound (~304MB traffic),
// bf16 WMMA keeps the 17.2 GFLOP of matmul far under the HBM floor.
// v2: use native __bf16 casts (RNE) so ISel can emit hardware cvt (packed)
//     instead of the 3-VALU-op manual rounding sequence per element.

typedef __attribute__((ext_vector_type(16))) __bf16 v16bf;
typedef __attribute__((ext_vector_type(8)))  float  v8f;

#define H_TOT 16
#define J_TOT 2048
#define K_TOT 2048
#define M_DIM 64

union BF16Frag {
    __bf16 b[16];
    v16bf  v;
};

// convert 8 consecutive f32 (16B-aligned) to bf16 with scale
__device__ __forceinline__ void cvt8_bf16(const float* __restrict__ s, float scale,
                                          __bf16* __restrict__ d) {
    float4 x = ((const float4*)s)[0];
    float4 y = ((const float4*)s)[1];
    d[0] = (__bf16)(x.x * scale); d[1] = (__bf16)(x.y * scale);
    d[2] = (__bf16)(x.z * scale); d[3] = (__bf16)(x.w * scale);
    d[4] = (__bf16)(y.x * scale); d[5] = (__bf16)(y.y * scale);
    d[6] = (__bf16)(y.z * scale); d[7] = (__bf16)(y.w * scale);
}

__global__ __launch_bounds__(256)
void KeyValAttention_65867618451509_kernel(const float* __restrict__ Q,
                                           const float* __restrict__ Km,
                                           const float* __restrict__ V,
                                           const int*   __restrict__ Mask,
                                           float* __restrict__ Out,
                                           float* __restrict__ Attn)
{
    __shared__ __align__(16) float sS[16][K_TOT];     // 128 KB raw scores -> masked p
    __shared__ __align__(16) float sOut[16][M_DIM];   // 4 KB  cross-wave out accum
    __shared__ float sRed[16][16];
    __shared__ float sMax[16];
    __shared__ float sInv[16];

    const int tid  = threadIdx.x;
    const int lane = tid & 31;
    const int wave = tid >> 5;
    const int h    = blockIdx.y;
    const int j0   = blockIdx.x * 16;

    const int halfsel = (lane < 16) ? 0 : 1;
    const int ln      = lane & 15;
    const float scale = 0.125f;  // 1/sqrt(64)

    const float* Qh = Q  + ((size_t)h * J_TOT + j0) * M_DIM;
    const float* Kh = Km + (size_t)h * K_TOT * M_DIM;
    const float* Vh = V  + (size_t)h * K_TOT * M_DIM;

    // zero the cross-wave output accumulator
    {
        float* o = &sOut[0][0];
        for (int i = tid; i < 16 * M_DIM; i += 256) o[i] = 0.0f;
    }

    // ---------------- Phase 1: S = (Q*scale) @ K^T via bf16 WMMA ----------------
    // A fragment (16x32 bf16): lane row = ln; lanes<16 k:{0..7,16..23}, lanes>=16 k:{8..15,24..31}
    BF16Frag a0, a1;
    {
        const float* qrow = Qh + (size_t)ln * M_DIM;
        const int off = halfsel ? 8 : 0;
        cvt8_bf16(qrow + 0  + off,      scale, a0.b);
        cvt8_bf16(qrow + 0  + off + 16, scale, a0.b + 8);
        cvt8_bf16(qrow + 32 + off,      scale, a1.b);
        cvt8_bf16(qrow + 32 + off + 16, scale, a1.b + 8);
    }

    for (int ct = wave; ct < K_TOT / 16; ct += 8) {
        const int k0 = ct * 16;
        // B fragment (32x16 bf16): lane col = key (k0+ln); lanes<16 k(m) {0..7,16..23}, lanes>=16 {8..15,24..31}
        const float* krow = Kh + (size_t)(k0 + ln) * M_DIM;
        const int ms = halfsel ? 8 : 0;
        BF16Frag b0, b1;
        cvt8_bf16(krow + ms,          1.0f, b0.b);
        cvt8_bf16(krow + ms + 16,     1.0f, b0.b + 8);
        cvt8_bf16(krow + 32 + ms,     1.0f, b1.b);
        cvt8_bf16(krow + 32 + ms + 16, 1.0f, b1.b + 8);

        v8f acc = {};
        acc = __builtin_amdgcn_wmma_f32_16x16x32_bf16(false, a0.v, false, b0.v,
                                                      (short)0, acc, false, false);
        acc = __builtin_amdgcn_wmma_f32_16x16x32_bf16(false, a1.v, false, b1.v,
                                                      (short)0, acc, false, false);
        // D layout: VGPR v -> row v (lanes 0-15) / v+8 (lanes 16-31), col = ln
        #pragma unroll
        for (int v = 0; v < 8; ++v)
            sS[v + (halfsel ? 8 : 0)][k0 + ln] = acc[v];
    }
    __syncthreads();

    // ---------------- Phase 2: softmax row max & sum (pre-mask, per reference) ----
    {
        const int row = tid >> 4, seg = tid & 15;
        const float4* sp = (const float4*)&sS[row][seg * 128];
        float m = -3.402823466e38f;
        #pragma unroll 4
        for (int i = 0; i < 32; ++i) {
            float4 s = sp[i];
            m = fmaxf(m, fmaxf(fmaxf(s.x, s.y), fmaxf(s.z, s.w)));
        }
        sRed[row][seg] = m;
    }
    __syncthreads();
    if (tid < 16) {
        float m = sRed[tid][0];
        #pragma unroll
        for (int i = 1; i < 16; ++i) m = fmaxf(m, sRed[tid][i]);
        sMax[tid] = m;
    }
    __syncthreads();
    {
        const int row = tid >> 4, seg = tid & 15;
        const float mx = sMax[row];
        const float4* sp = (const float4*)&sS[row][seg * 128];
        float sum = 0.0f;
        #pragma unroll 4
        for (int i = 0; i < 32; ++i) {
            float4 s = sp[i];
            sum += __expf(s.x - mx) + __expf(s.y - mx) +
                   __expf(s.z - mx) + __expf(s.w - mx);
        }
        sRed[row][seg] = sum;
    }
    __syncthreads();
    if (tid < 16) {
        float s = 0.0f;
        #pragma unroll
        for (int i = 0; i < 16; ++i) s += sRed[tid][i];
        sInv[tid] = 1.0f / s;
    }
    __syncthreads();

    // ---------------- Phase 3a: p = softmax, post-softmax mask fill, stream attn --
    {
        float*     attnBase = Attn + ((size_t)h * J_TOT + j0) * K_TOT;
        const int* maskBase = Mask + (size_t)j0 * K_TOT;   // mask broadcast over H
        #pragma unroll 2
        for (int it = 0; it < 32; ++it) {
            const int idx = tid + it * 256;      // float4 index in [0, 8192)
            const int row = idx >> 9;            // 512 float4 per row
            const int col = (idx & 511) * 4;
            float4 s = *(const float4*)&sS[row][col];
            const float mx = sMax[row], ri = sInv[row];
            float4 p;
            p.x = __expf(s.x - mx) * ri;
            p.y = __expf(s.y - mx) * ri;
            p.z = __expf(s.z - mx) * ri;
            p.w = __expf(s.w - mx) * ri;
            int4 mk = *(const int4*)(maskBase + (size_t)row * K_TOT + col);
            p.x = mk.x ? p.x : 1e-9f;
            p.y = mk.y ? p.y : 1e-9f;
            p.z = mk.z ? p.z : 1e-9f;
            p.w = mk.w ? p.w : 1e-9f;
            *(float4*)(attnBase + (size_t)row * K_TOT + col) = p;   // coalesced HBM store
            *(float4*)&sS[row][col] = p;                            // masked p for P@V
        }
    }
    __syncthreads();

    // ---------------- Phase 3b: out = P @ V via bf16 WMMA ------------------------
    {
        v8f acc[4] = {};
        const int off1 = halfsel ? 8 : 0;
        const int klo  = halfsel ? 16 : 0;
        for (int c = wave; c < K_TOT / 32; c += 8) {
            const int kk0 = c * 32;
            // A fragment: masked P rows from LDS (contraction dim = key, contiguous)
            BF16Frag pa;
            const float* prow = &sS[ln][kk0];
            cvt8_bf16(prow + klo,      1.0f, pa.b);
            cvt8_bf16(prow + klo + 8,  1.0f, pa.b + 8);
            #pragma unroll
            for (int nt = 0; nt < 4; ++nt) {
                // B fragment: V[k, m], lane col m = nt*16+ln, 16 consecutive keys
                BF16Frag vb;
                const float* vcol = Vh + (size_t)(kk0 + klo) * M_DIM + nt * 16 + ln;
                #pragma unroll
                for (int i = 0; i < 16; ++i)
                    vb.b[i] = (__bf16)vcol[(size_t)i * M_DIM];
                acc[nt] = __builtin_amdgcn_wmma_f32_16x16x32_bf16(false, pa.v, false, vb.v,
                                                                  (short)0, acc[nt], false, false);
            }
        }
        // cross-wave reduction into sOut
        #pragma unroll
        for (int nt = 0; nt < 4; ++nt)
            #pragma unroll
            for (int v = 0; v < 8; ++v)
                atomicAdd(&sOut[v + (halfsel ? 8 : 0)][nt * 16 + ln], acc[nt][v]);
    }
    __syncthreads();

    // ---------------- write out tile (16x64 f32) ---------------------------------
    {
        const int row = tid >> 4;
        const int col = (tid & 15) * 4;
        float4 o = *(const float4*)&sOut[row][col];
        *(float4*)(Out + ((size_t)h * J_TOT + j0 + row) * M_DIM + col) = o;
    }
}

extern "C" void kernel_launch(void* const* d_in, const int* in_sizes, int n_in,
                              void* d_out, int out_size, void* d_ws, size_t ws_size,
                              hipStream_t stream) {
    (void)in_sizes; (void)n_in; (void)out_size; (void)d_ws; (void)ws_size;
    const float* Q    = (const float*)d_in[0];
    const float* K    = (const float*)d_in[1];
    const float* V    = (const float*)d_in[2];
    const int*   Mask = (const int*)d_in[3];
    float* Out  = (float*)d_out;
    float* Attn = Out + (size_t)H_TOT * J_TOT * M_DIM;   // outputs concatenated: out, attn

    dim3 grid(J_TOT / 16, H_TOT);   // 128 x 16 = 2048 workgroups
    KeyValAttention_65867618451509_kernel<<<grid, 256, 0, stream>>>(Q, K, V, Mask, Out, Attn);
}
